// EEGMambaForEval_43714177139204
// MI455X (gfx1250) — matrix operand order repaired and checked by reading, not compiled, not run on hardware
//
#include <hip/hip_runtime.h>
#include <hip/hip_bf16.h>
#include <math.h>

// ---------------------------------------------------------------------------
// EEGMamba forward for MI455X (gfx1250). bf16 WMMA for all GEMMs, register-
// resident selective scan, fused elementwise kernels.
// ---------------------------------------------------------------------------

typedef __attribute__((ext_vector_type(16))) __bf16        v16bf;
typedef __attribute__((ext_vector_type(8)))  float         v8f;
typedef __attribute__((ext_vector_type(8)))  unsigned int  v8u;

#define D_MODEL  768
#define D_INNER  1536
#define DT_RANK_ 48
#define DT_PAD   64
#define N_STATE  16
#define NB       4
#define SEQ      512
#define NTOK     (NB * SEQ)   // 2048
#define NLAYER   4
#define NC       256          // input channels

__device__ __forceinline__ unsigned short f32_to_bf16(float f) {
  unsigned int u = __builtin_bit_cast(unsigned int, f);
  unsigned int r = u + 0x7FFFu + ((u >> 16) & 1u);   // round-to-nearest-even
  return (unsigned short)(r >> 16);
}
__device__ __forceinline__ float siluf(float x) { return x / (1.f + __expf(-x)); }
__device__ __forceinline__ float softplusf(float x) { return x > 20.f ? x : log1pf(__expf(x)); }

// Force global (addrspace 1) memory ops: avoids flat_load (which ticks DScnt
// and does per-lane aperture checks) when pointer provenance is obscured.
__device__ __forceinline__ unsigned int ld_g32(const unsigned short* p) {
  return *(const __attribute__((address_space(1))) unsigned int*)p;
}
__device__ __forceinline__ void st_g32(float* p, float v) {
  *(__attribute__((address_space(1))) float*)p = v;
}

// ---------------------------------------------------------------------------
// bf16 GEMM:  C[M x N] = A[M x K] * Bt[N x K]^T   (f32 accumulate)
// K multiple of 32, M multiple of 256.
// Wave -> 32(M) x 16*NSUB(N) tile: 2 A-fragments x NSUB B-fragments,
// 2*NSUB WMMA per K-step (each B fragment reused twice in registers).
// Block = 8 waves = 256(M) x 16*NSUB(N).  EPI: 0=store, 1=+bias, 2=softplus.
// NSUB/EPI compile-time -> straight-line control flow, no spills, EXEC all-1s
// for WMMA.  Fragment layouts per CDNA5 ISA 7.12.2 (wave32).
// ---------------------------------------------------------------------------
template <int NSUB, int EPI>
__global__ __launch_bounds__(256)
void gemm_bf16_wmma(const unsigned short* __restrict__ A,
                    const unsigned short* __restrict__ Bt,
                    float* __restrict__ C,
                    const float* __restrict__ bias,
                    int M, int N, int K, int nBase)
{
  const int lane  = threadIdx.x & 31;
  const int wave  = threadIdx.x >> 5;
  const int half  = lane >> 4;     // 0: lanes 0-15, 1: lanes 16-31
  const int lm    = lane & 15;
  const int mtile = blockIdx.y * 8 + wave;      // 32 rows per wave
  const int n0    = nBase + blockIdx.x * (16 * NSUB);

  v8f acc[2][NSUB] = {};

  // A fragment rows: m0 + lm (frag 0), m0 + 16 + lm (frag 1)
  const unsigned short* pa0 = A + (size_t)(mtile * 32 + lm) * K + half * 8;
  const unsigned short* pa1 = pa0 + (size_t)16 * K;
  const unsigned short* pb[NSUB];
#pragma unroll
  for (int s = 0; s < NSUB; ++s)
    pb[s] = Bt + (size_t)(n0 + s * 16 + lm) * K + half * 8;

  for (int kb = 0; kb < K; kb += 32) {
    // A fragments: 16x32 bf16.  VGPR j, half h -> K pair ((j<4)?0:16)+8h+2(j%4)
    v8u au0, au1;
#pragma unroll
    for (int j = 0; j < 8; ++j) {
      const int koff = ((j & 4) ? 16 : 0) + (j & 3) * 2;
      au0[j] = ld_g32(pa0 + koff);
      au1[j] = ld_g32(pa1 + koff);
    }
    v16bf a0 = __builtin_bit_cast(v16bf, au0);
    v16bf a1 = __builtin_bit_cast(v16bf, au1);

#pragma unroll
    for (int s = 0; s < NSUB; ++s) {
      // B fragment: 32x16 bf16. col n = lm; VGPR j, half h -> K pair ((j<4)?0:16)+8h+2(j%4)
      v8u bu;
#pragma unroll
      for (int j = 0; j < 8; ++j)
        bu[j] = ld_g32(pb[s] + ((j & 4) ? 16 : 0) + (j & 3) * 2);
      v16bf b = __builtin_bit_cast(v16bf, bu);
      acc[0][s] = __builtin_amdgcn_wmma_f32_16x16x32_bf16(
          false, a0, false, b, (short)0, acc[0][s], false, false);
      acc[1][s] = __builtin_amdgcn_wmma_f32_16x16x32_bf16(
          false, a1, false, b, (short)0, acc[1][s], false, false);
      pb[s] += 32;
    }
    pa0 += 32;
    pa1 += 32;
  }

#pragma unroll
  for (int h = 0; h < 2; ++h) {
#pragma unroll
    for (int s = 0; s < NSUB; ++s) {
      const int n = n0 + s * 16 + lm;
      float bv = (EPI >= 1) ? bias[n] : 0.f;
#pragma unroll
      for (int r = 0; r < 8; ++r) {         // C layout: VGPR r -> M = r + 8*half
        const int mm = mtile * 32 + h * 16 + r + 8 * half;
        float v = acc[h][s][r] + bv;
        if (EPI == 2) v = softplusf(v);
        st_g32(C + (size_t)mm * N + n, v);
      }
    }
  }
}

// ---------------------------------------------------------------------------
// Elementwise / helper kernels
// ---------------------------------------------------------------------------
__global__ void convert_bf16_kernel(const float* __restrict__ src,
                                    unsigned short* __restrict__ dst, int n) {
  int i = blockIdx.x * 256 + threadIdx.x;
  if (i < n) dst[i] = f32_to_bf16(src[i]);
}

// x:(B,C,T) f32 -> xbf:(B*T, C) bf16 (token-major for GEMM)
__global__ void transpose_x_bf16(const float* __restrict__ x,
                                 unsigned short* __restrict__ xbf) {
  int idx = blockIdx.x * 256 + threadIdx.x;
  if (idx >= NTOK * NC) return;
  int c = idx % NC; int tok = idx / NC;
  int t = tok % SEQ, b = tok / SEQ;
  xbf[idx] = f32_to_bf16(x[((size_t)(b * NC + c)) * SEQ + t]);
}

// pad rows of 48 f32 -> rows of 64 bf16 (zero fill)
__global__ void pad48to64_bf16(const float* __restrict__ src,
                               unsigned short* __restrict__ dst, int rows) {
  int i = blockIdx.x * 256 + threadIdx.x;
  if (i >= rows * DT_PAD) return;
  int r = i / DT_PAD, j = i % DT_PAD;
  dst[i] = (j < DT_RANK_) ? f32_to_bf16(src[r * DT_RANK_ + j]) : (unsigned short)0;
}

// dt slice of xp (stride 80, cols 0..47) -> padded bf16 (2048 x 64)
__global__ void split_dt_kernel(const float* __restrict__ xp,
                                unsigned short* __restrict__ dtpad) {
  int i = blockIdx.x * 256 + threadIdx.x;
  if (i >= NTOK * DT_PAD) return;
  int r = i / DT_PAD, j = i % DT_PAD;
  dtpad[i] = (j < DT_RANK_) ? f32_to_bf16(xp[r * 80 + j]) : (unsigned short)0;
}

// Fused residual add + RMSNorm.  Writes new residual, normed bf16 (+opt f32).
__global__ void rmsnorm_kernel(const float* __restrict__ hid,
                               float* __restrict__ res,
                               const float* __restrict__ w,
                               unsigned short* __restrict__ outBf,
                               float* __restrict__ outF32,
                               int addRes)
{
  const int tok = blockIdx.x;
  const int tid = threadIdx.x;
  __shared__ float sred[256];
  float v[3];
  float ss = 0.f;
#pragma unroll
  for (int i = 0; i < 3; ++i) {
    int j = tid + i * 256;
    float r = hid[(size_t)tok * D_MODEL + j];
    if (addRes) r += res[(size_t)tok * D_MODEL + j];
    v[i] = r;
    ss += r * r;
  }
  sred[tid] = ss;
  __syncthreads();
  for (int off = 128; off > 0; off >>= 1) {
    if (tid < off) sred[tid] += sred[tid + off];
    __syncthreads();
  }
  float rstd = rsqrtf(sred[0] / (float)D_MODEL + 1e-5f);
#pragma unroll
  for (int i = 0; i < 3; ++i) {
    int j = tid + i * 256;
    float r = v[i];
    res[(size_t)tok * D_MODEL + j] = r;
    float nv = r * rstd * w[j];
    outBf[(size_t)tok * D_MODEL + j] = f32_to_bf16(nv);
    if (outF32) outF32[(size_t)tok * D_MODEL + j] = nv;
  }
}

// Causal depthwise conv (K=4) over u-half of xz, + bias, SiLU. f32 + bf16 out.
__global__ void conv_silu_kernel(const float* __restrict__ xz,
                                 const float* __restrict__ cw,
                                 const float* __restrict__ cb,
                                 float* __restrict__ u,
                                 unsigned short* __restrict__ ubf)
{
  int idx = blockIdx.x * 256 + threadIdx.x;
  if (idx >= NTOK * D_INNER) return;
  int d = idx % D_INNER;
  int tok = idx / D_INNER;
  int t = tok % SEQ, b = tok / SEQ;
  float acc = cb[d];
#pragma unroll
  for (int k = 0; k < 4; ++k) {
    int tt = t - 3 + k;
    if (tt >= 0)
      acc += xz[((size_t)(b * SEQ + tt)) * (2 * D_INNER) + d] * cw[d * 4 + k];
  }
  float s = siluf(acc);
  u[idx]   = s;
  ubf[idx] = f32_to_bf16(s);
}

// Selective scan: one lane per (b, d); 16 states + 16 A values in VGPRs.
// B/C (shared per (b,t)) staged through LDS.  Fuses +u*D and *silu(z).
__global__ void mamba_scan_kernel(const float* __restrict__ delta,
                                  const float* __restrict__ u,
                                  const float* __restrict__ xz,
                                  const float* __restrict__ xp,
                                  const float* __restrict__ alog,
                                  const float* __restrict__ dpar,
                                  unsigned short* __restrict__ ybf)
{
  const int b = blockIdx.x / (D_INNER / 256);
  const int d = (blockIdx.x % (D_INNER / 256)) * 256 + threadIdx.x;
  float Aa[N_STATE];
#pragma unroll
  for (int n = 0; n < N_STATE; ++n) Aa[n] = -__expf(alog[d * N_STATE + n]);
  const float Dd = dpar[d];
  float st[N_STATE];
#pragma unroll
  for (int n = 0; n < N_STATE; ++n) st[n] = 0.f;
  __shared__ float sh[32];     // Bm[0..15], Cm[0..15]

  for (int t = 0; t < SEQ; ++t) {
    int tok = b * SEQ + t;
    __syncthreads();
    if (threadIdx.x < 32) sh[threadIdx.x] = xp[(size_t)tok * 80 + DT_RANK_ + threadIdx.x];
    __syncthreads();
    float dl = delta[(size_t)tok * D_INNER + d];
    float ul = u[(size_t)tok * D_INNER + d];
    float zl = xz[(size_t)tok * (2 * D_INNER) + D_INNER + d];
    float du = dl * ul;
    float y = 0.f;
#pragma unroll
    for (int n = 0; n < N_STATE; ++n) {
      st[n] = __expf(dl * Aa[n]) * st[n] + du * sh[n];
      y += st[n] * sh[16 + n];
    }
    y = (y + ul * Dd) * siluf(zl);
    ybf[(size_t)tok * D_INNER + d] = f32_to_bf16(y);
  }
}

// Mean-pool over T
__global__ void pool_kernel(const float* __restrict__ nf, float* __restrict__ pooled) {
  int idx = blockIdx.x * 256 + threadIdx.x;
  if (idx >= NB * D_MODEL) return;
  int b = idx / D_MODEL, j = idx % D_MODEL;
  float s = 0.f;
  for (int t = 0; t < SEQ; ++t) s += nf[((size_t)(b * SEQ + t)) * D_MODEL + j];
  pooled[idx] = s * (1.f / (float)SEQ);
}

// Tiny classifier head: 40 dot products of length 768
__global__ void logits_kernel(const float* __restrict__ pooled,
                              const float* __restrict__ clsw,
                              const float* __restrict__ clsb,
                              float* __restrict__ out) {
  int i = threadIdx.x;
  if (i >= NB * 10) return;
  int b = i / 10, c = i % 10;
  float s = clsb[c];
  for (int j = 0; j < D_MODEL; ++j) s += pooled[b * D_MODEL + j] * clsw[c * D_MODEL + j];
  out[i] = s;
}

// ---------------------------------------------------------------------------
// Host orchestration
// ---------------------------------------------------------------------------
extern "C" void kernel_launch(void* const* d_in, const int* in_sizes, int n_in,
                              void* d_out, int out_size, void* d_ws, size_t ws_size,
                              hipStream_t stream) {
  const float* x        = (const float*)d_in[0];
  const float* in_proj  = (const float*)d_in[1];
  const float* conv_w   = (const float*)d_in[2];
  const float* conv_b   = (const float*)d_in[3];
  const float* x_proj   = (const float*)d_in[4];
  const float* dt_projw = (const float*)d_in[5];
  const float* dt_projb = (const float*)d_in[6];
  const float* A_log    = (const float*)d_in[7];
  const float* D_param  = (const float*)d_in[8];
  const float* out_proj = (const float*)d_in[9];
  const float* norm_w   = (const float*)d_in[10];
  const float* norm_f_w = (const float*)d_in[11];
  const float* inp_w    = (const float*)d_in[12];
  const float* inp_b    = (const float*)d_in[13];
  const float* cls_w    = (const float*)d_in[14];
  const float* cls_b    = (const float*)d_in[15];
  float* out = (float*)d_out;

  char* ws = (char*)d_ws;
  size_t off = 0;
  auto alloc = [&](size_t bytes) -> void* {
    void* p = ws + off; off += (bytes + 255) & ~(size_t)255; return p;
  };

  unsigned short* xbf      = (unsigned short*)alloc((size_t)NTOK * NC * 2);
  unsigned short* inpw_bf  = (unsigned short*)alloc((size_t)D_MODEL * NC * 2);
  unsigned short* ipw_bf   = (unsigned short*)alloc((size_t)2 * D_INNER * D_MODEL * 2);
  unsigned short* xpw_bf   = (unsigned short*)alloc((size_t)80 * D_INNER * 2);
  unsigned short* dtw_bf   = (unsigned short*)alloc((size_t)D_INNER * DT_PAD * 2);
  unsigned short* opw_bf   = (unsigned short*)alloc((size_t)D_MODEL * D_INNER * 2);
  float*          hiddenB  = (float*)alloc((size_t)NTOK * D_MODEL * 4);
  float*          resB     = (float*)alloc((size_t)NTOK * D_MODEL * 4);
  unsigned short* normedBf = (unsigned short*)alloc((size_t)NTOK * D_MODEL * 2);
  float*          normedF  = (float*)alloc((size_t)NTOK * D_MODEL * 4);
  float*          xz       = (float*)alloc((size_t)NTOK * 2 * D_INNER * 4);
  float*          uB       = (float*)alloc((size_t)NTOK * D_INNER * 4);
  unsigned short* ubf      = (unsigned short*)alloc((size_t)NTOK * D_INNER * 2);
  float*          xp       = (float*)alloc((size_t)NTOK * 80 * 4);
  unsigned short* dtpad    = (unsigned short*)alloc((size_t)NTOK * DT_PAD * 2);
  float*          deltaB   = (float*)alloc((size_t)NTOK * D_INNER * 4);
  unsigned short* ybf      = (unsigned short*)alloc((size_t)NTOK * D_INNER * 2);
  float*          pooled   = (float*)alloc((size_t)NB * D_MODEL * 4);
  (void)ws_size; (void)in_sizes; (void)n_in; (void)out_size;

  auto cdiv = [](int a, int b) { return (a + b - 1) / b; };

  // GEMM dispatcher: main NSUB=4 tiles + remainder tiles, all compile-time.
  auto launch_gemm = [&](const unsigned short* A, const unsigned short* Bt, float* C,
                         const float* bias, int M, int N, int K, int epi) {
    const int nMain = N / 64;
    const int rem   = (N % 64) / 16;
    dim3 blk(256);
    if (nMain > 0) {
      dim3 g(nMain, M / 256);
      if (epi == 0)      gemm_bf16_wmma<4, 0><<<g, blk, 0, stream>>>(A, Bt, C, bias, M, N, K, 0);
      else if (epi == 1) gemm_bf16_wmma<4, 1><<<g, blk, 0, stream>>>(A, Bt, C, bias, M, N, K, 0);
      else               gemm_bf16_wmma<4, 2><<<g, blk, 0, stream>>>(A, Bt, C, bias, M, N, K, 0);
    }
    if (rem > 0) {
      dim3 g(1, M / 256);
      const int nBase = nMain * 64;
      if (epi == 0) {
        if (rem == 1)      gemm_bf16_wmma<1, 0><<<g, blk, 0, stream>>>(A, Bt, C, bias, M, N, K, nBase);
        else if (rem == 2) gemm_bf16_wmma<2, 0><<<g, blk, 0, stream>>>(A, Bt, C, bias, M, N, K, nBase);
        else               gemm_bf16_wmma<3, 0><<<g, blk, 0, stream>>>(A, Bt, C, bias, M, N, K, nBase);
      } else if (epi == 1) {
        if (rem == 1)      gemm_bf16_wmma<1, 1><<<g, blk, 0, stream>>>(A, Bt, C, bias, M, N, K, nBase);
        else if (rem == 2) gemm_bf16_wmma<2, 1><<<g, blk, 0, stream>>>(A, Bt, C, bias, M, N, K, nBase);
        else               gemm_bf16_wmma<3, 1><<<g, blk, 0, stream>>>(A, Bt, C, bias, M, N, K, nBase);
      } else {
        if (rem == 1)      gemm_bf16_wmma<1, 2><<<g, blk, 0, stream>>>(A, Bt, C, bias, M, N, K, nBase);
        else if (rem == 2) gemm_bf16_wmma<2, 2><<<g, blk, 0, stream>>>(A, Bt, C, bias, M, N, K, nBase);
        else               gemm_bf16_wmma<3, 2><<<g, blk, 0, stream>>>(A, Bt, C, bias, M, N, K, nBase);
      }
    }
  };

  // --- input projection: h = x^T @ inp_w^T + inp_b ---
  transpose_x_bf16<<<cdiv(NTOK * NC, 256), 256, 0, stream>>>(x, xbf);
  convert_bf16_kernel<<<cdiv(D_MODEL * NC, 256), 256, 0, stream>>>(inp_w, inpw_bf, D_MODEL * NC);
  launch_gemm(xbf, inpw_bf, hiddenB, inp_b, NTOK, D_MODEL, NC, 1);

  for (int l = 0; l < NLAYER; ++l) {
    const int IPW = 2 * D_INNER * D_MODEL;   // 3072*768
    const int XPW = 80 * D_INNER;
    const int OPW = D_MODEL * D_INNER;
    convert_bf16_kernel<<<cdiv(IPW, 256), 256, 0, stream>>>(in_proj + (size_t)l * IPW, ipw_bf, IPW);
    convert_bf16_kernel<<<cdiv(XPW, 256), 256, 0, stream>>>(x_proj + (size_t)l * XPW, xpw_bf, XPW);
    pad48to64_bf16<<<cdiv(D_INNER * DT_PAD, 256), 256, 0, stream>>>(
        dt_projw + (size_t)l * D_INNER * DT_RANK_, dtw_bf, D_INNER);
    convert_bf16_kernel<<<cdiv(OPW, 256), 256, 0, stream>>>(out_proj + (size_t)l * OPW, opw_bf, OPW);

    // residual += hidden; normed = rmsnorm(residual) -> bf16
    rmsnorm_kernel<<<NTOK, 256, 0, stream>>>(hiddenB, resB, norm_w + (size_t)l * D_MODEL,
                                             normedBf, nullptr, l > 0 ? 1 : 0);
    // xz = normed @ in_proj^T
    launch_gemm(normedBf, ipw_bf, xz, nullptr, NTOK, 2 * D_INNER, D_MODEL, 0);
    // u = silu(causal_dw_conv(u_half) + b)
    conv_silu_kernel<<<cdiv(NTOK * D_INNER, 256), 256, 0, stream>>>(
        xz, conv_w + (size_t)l * D_INNER * 4, conv_b + (size_t)l * D_INNER, uB, ubf);
    // xp = u @ x_proj^T   (N = 80 -> NSUB=4 block + NSUB=1 remainder)
    launch_gemm(ubf, xpw_bf, xp, nullptr, NTOK, 80, D_INNER, 0);
    // dt (padded K=64)
    split_dt_kernel<<<cdiv(NTOK * DT_PAD, 256), 256, 0, stream>>>(xp, dtpad);
    // delta = softplus(dt @ dt_proj^T + dt_b)
    launch_gemm(dtpad, dtw_bf, deltaB, dt_projb + (size_t)l * D_INNER, NTOK, D_INNER, DT_PAD, 2);
    // selective scan (fused +u*D, *silu(z)) -> y bf16
    mamba_scan_kernel<<<NB * (D_INNER / 256), 256, 0, stream>>>(
        deltaB, uB, xz, xp, A_log + (size_t)l * D_INNER * N_STATE,
        D_param + (size_t)l * D_INNER, ybf);
    // hidden = y @ out_proj^T
    launch_gemm(ybf, opw_bf, hiddenB, nullptr, NTOK, D_MODEL, D_INNER, 0);
  }

  // final residual + rmsnorm (f32), pool, classify
  rmsnorm_kernel<<<NTOK, 256, 0, stream>>>(hiddenB, resB, norm_f_w, normedBf, normedF, 1);
  pool_kernel<<<cdiv(NB * D_MODEL, 256), 256, 0, stream>>>(normedF, pooled);
  logits_kernel<<<1, 64, 0, stream>>>(pooled, cls_w, cls_b, out);
}
